// MyModel_49898930045109
// MI455X (gfx1250) — compile-verified, matrix operand order checked by reading, not run on hardware
//
#include <hip/hip_runtime.h>
#include <hip/hip_bf16.h>

typedef _Float16 v16h __attribute__((ext_vector_type(16)));
typedef _Float16 v8h  __attribute__((ext_vector_type(8)));
typedef float    v8f  __attribute__((ext_vector_type(8)));

#define Bn   2
#define Sn   2048
#define Hn   32
#define Dn   96
#define HIDn 3072
#define NQKV (3*HIDn)   // 9216
#define Mtok (Bn*Sn)    // 4096

// Load a 16-half WMMA fragment from two contiguous 16-byte chunks (p, p+16 halves).
// Matches the CDNA5 16-bit A/B layout: lanes 0-15 take K {k0..k0+7, k0+16..k0+23},
// lanes 16-31 take K {k0+8..k0+15, k0+24..k0+31} (caller adds the +8 lane offset).
__device__ __forceinline__ v16h frag2(const _Float16* p) {
    v8h lo = *(const v8h*)p;
    v8h hi = *(const v8h*)(p + 16);
    return __builtin_shufflevector(lo, hi, 0,1,2,3,4,5,6,7,8,9,10,11,12,13,14,15);
}

__device__ __forceinline__ v8f wmma_f16(v16h a, v16h b, v8f c) {
    return __builtin_amdgcn_wmma_f32_16x16x32_f16(false, a, false, b, (short)0, c, false, false);
}

// ---------------------------------------------------------------- f32 -> f16
__global__ void cvt_kernel(const float* __restrict__ in, _Float16* __restrict__ out,
                           long long n) {
    long long i = (long long)blockIdx.x * blockDim.x + threadIdx.x;
    long long stride = (long long)gridDim.x * blockDim.x;
    for (; i < n; i += stride) out[i] = (_Float16)in[i];
}

// ------------------------------------------------ GEMM: C[M,N] = A[M,K] * B[N,K]^T
// Block = 256 threads = 8 waves; block tile 128x128; wave tile 64x32 (4x2 WMMA).
template<int OUTF32>
__global__ __launch_bounds__(256, 1)
void gemm_nt_kernel(const _Float16* __restrict__ A,
                    const _Float16* __restrict__ Bw,
                    void* __restrict__ outp,
                    int M, int N, int K) {
    const int lane = threadIdx.x & 31;
    const int wave = threadIdx.x >> 5;
    const int wm   = wave >> 2;          // 0..1
    const int wn   = wave & 3;           // 0..3
    const int l15  = lane & 15;
    const int koff = (lane < 16) ? 0 : 8;
    const int rbase = blockIdx.y * 128 + wm * 64;
    const int cbase = blockIdx.x * 128 + wn * 32;

    v8f vzero = {};
    v8f acc[4][2];
#pragma unroll
    for (int mt = 0; mt < 4; ++mt)
#pragma unroll
        for (int nt = 0; nt < 2; ++nt) acc[mt][nt] = vzero;

    const _Float16* Arow[4];
    const _Float16* Brow[2];
#pragma unroll
    for (int mt = 0; mt < 4; ++mt)
        Arow[mt] = A + (size_t)(rbase + mt*16 + l15) * K + koff;
#pragma unroll
    for (int nt = 0; nt < 2; ++nt)
        Brow[nt] = Bw + (size_t)(cbase + nt*16 + l15) * K + koff;

    for (int k0 = 0; k0 < K; k0 += 32) {
        // distance-scheduled prefetch (global_prefetch_b8): 512B = 8 k-steps ahead
#pragma unroll
        for (int mt = 0; mt < 4; ++mt) __builtin_prefetch(Arow[mt] + k0 + 256, 0, 3);
#pragma unroll
        for (int nt = 0; nt < 2; ++nt) __builtin_prefetch(Brow[nt] + k0 + 256, 0, 3);

        v16h af[4], bf[2];
#pragma unroll
        for (int mt = 0; mt < 4; ++mt) af[mt] = frag2(Arow[mt] + k0);
#pragma unroll
        for (int nt = 0; nt < 2; ++nt) bf[nt] = frag2(Brow[nt] + k0);
#pragma unroll
        for (int mt = 0; mt < 4; ++mt)
#pragma unroll
            for (int nt = 0; nt < 2; ++nt)
                acc[mt][nt] = wmma_f16(af[mt], bf[nt], acc[mt][nt]);
    }

    const int mloc = (lane >> 4) << 3;   // C layout: VGPR r -> row r (+8 for hi half)
#pragma unroll
    for (int mt = 0; mt < 4; ++mt)
#pragma unroll
        for (int nt = 0; nt < 2; ++nt)
#pragma unroll
            for (int r = 0; r < 8; ++r) {
                int m = rbase + mt*16 + r + mloc;
                int n = cbase + nt*16 + l15;
                float v = acc[mt][nt][r];
                if (OUTF32) ((float*)outp)[(size_t)m * N + n] = v;
                else        ((_Float16*)outp)[(size_t)m * N + n] = (_Float16)v;
            }
}

// ------------------------------------------------ RoPE + reshape to attention layouts
// Q (pre-scaled by 1/sqrt(D)) and K -> [B,H,S,D] f16 ; V -> [B,H,D,S] f16 transposed
__global__ void rope_kernel(const _Float16* __restrict__ qkv,
                            const float* __restrict__ cosb,
                            const float* __restrict__ sinb,
                            _Float16* __restrict__ Qo,
                            _Float16* __restrict__ Ko,
                            _Float16* __restrict__ VTo) {
    int idx = blockIdx.x * blockDim.x + threadIdx.x;   // over B*S*H*D
    if (idx >= Bn*Sn*Hn*Dn) return;
    int d = idx % Dn;
    int h = (idx / Dn) % Hn;
    int s = (idx / (Dn*Hn)) % Sn;
    int b =  idx / (Dn*Hn*Sn);

    size_t rowoff = ((size_t)(b*Sn + s)) * NQKV + h * Dn;
    float q = (float)qkv[rowoff + d];
    float k = (float)qkv[rowoff + HIDn + d];
    float v = (float)qkv[rowoff + 2*HIDn + d];
    int   dp  = (d < Dn/2) ? (d + Dn/2) : (d - Dn/2);
    float sgn = (d < Dn/2) ? -1.0f : 1.0f;
    float qp = (float)qkv[rowoff + dp];
    float kp = (float)qkv[rowoff + HIDn + dp];
    float c  = cosb[((size_t)(b*Sn + s)) * Dn + d];
    float sv = sinb[((size_t)(b*Sn + s)) * Dn + d];

    const float qscale = 0.10206207262f;   // 1/sqrt(96), folded into Q here
    size_t bh = (size_t)(b*Hn + h);
    Qo[(bh*Sn + s)*Dn + d] = (_Float16)((q*c + sgn*qp*sv) * qscale);
    Ko[(bh*Sn + s)*Dn + d] = (_Float16)(k*c + sgn*kp*sv);
    VTo[(bh*Dn + d)*Sn + s] = (_Float16)v;
}

// ------------------------------------------------ Causal flash attention
// Transposed-score formulation: S^T = K * Q^T, so each lane owns ONE query column
// and its key scores live in-lane (8 VGPRs x 2 tiles; other key half in lane^16).
// Softmax needs only one shfl_xor(16) per reduction, and exp(S^T) registers ARE
// the A-fragment layout of P for the P*V WMMA -> no LDS round trip at all.
//
// One 32-key online-softmax step. MASKED=true only for the single tail iteration
// (kv0 is a multiple of 32, q0 of 16 -> only the last step can cross the diagonal).
template<bool MASKED>
__device__ __forceinline__ void attn_step(
    int kv0, int q0, int lane,
    const _Float16* __restrict__ Kb, const _Float16* __restrict__ Vb,
    const v16h (&qf)[3], v8f (&acc)[6],
    float& mcur, float& lcur) {
    const int l15  = lane & 15;
    const int koff = (lane < 16) ? 0 : 8;
    const int mloc = (lane >> 4) << 3;   // +8 for upper half (C-layout M offset)

    // prefetch next step's K/V tile rows toward WGP$/GL2
    __builtin_prefetch(Kb + (size_t)(kv0 + 32 + l15) * Dn, 0, 3);
    __builtin_prefetch(Vb + (size_t)(lane * 3) * Sn + kv0 + 32, 0, 3);

    // ---- S^T tiles: M = keys (VGPR index + half), N = queries (lane)
    v8f vz = {};
    v8f st[2]; st[0] = vz; st[1] = vz;
#pragma unroll
    for (int nt = 0; nt < 2; ++nt) {
        const _Float16* kr = Kb + (size_t)(kv0 + nt*16 + l15) * Dn + koff;
#pragma unroll
        for (int i = 0; i < 3; ++i) {
            v16h kf = frag2(kr + i * 32);          // A operand: 16 keys x 32 d
            st[nt] = wmma_f16(kf, qf[i], st[nt]);  // B operand: 32 d x 16 queries
        }
    }

    if (MASKED) {
        const int query = q0 + l15;
#pragma unroll
        for (int nt = 0; nt < 2; ++nt)
#pragma unroll
            for (int r = 0; r < 8; ++r) {
                int key = kv0 + nt*16 + r + mloc;
                st[nt][r] = (key <= query) ? st[nt][r] : -1e30f;
            }
    }

    // ---- per-query max: in-lane tree over 16 values + one cross-half exchange
    float mx = fmaxf(st[0][0], st[1][0]);
#pragma unroll
    for (int r = 1; r < 8; ++r)
        mx = fmaxf(mx, fmaxf(st[0][r], st[1][r]));
    mx = fmaxf(mx, __shfl_xor(mx, 16, 32));

    float mn    = fmaxf(mcur, mx);
    float alpha = __expf(mcur - mn);
    mcur = mn;

    // ---- P^T = exp(S^T - m): converts straight into the A fragment of P
    v16h pf;
    float sum = 0.0f;
#pragma unroll
    for (int nt = 0; nt < 2; ++nt)
#pragma unroll
        for (int r = 0; r < 8; ++r) {
            float p = __expf(st[nt][r] - mn);
            sum += p;
            pf[nt*8 + r] = (_Float16)p;
        }
    sum += __shfl_xor(sum, 16, 32);
    lcur = lcur * alpha + sum;

    // ---- broadcast per-query alpha (lane-resident) into the ctx C-row domain
    float abc[8];
#pragma unroll
    for (int r = 0; r < 8; ++r)
        abc[r] = __shfl(alpha, r + mloc, 32);
#pragma unroll
    for (int t = 0; t < 6; ++t)
#pragma unroll
        for (int r = 0; r < 8; ++r)
            acc[t][r] *= abc[r];

    // ---- ctx += P * V  (B frags from V^T: lane = d column, contiguous keys)
#pragma unroll
    for (int t = 0; t < 6; ++t) {
        const _Float16* vr = Vb + (size_t)(t*16 + l15) * Sn + kv0 + koff;
        v16h vf = frag2(vr);
        acc[t] = wmma_f16(pf, vf, acc[t]);
    }
}

// 256 threads = 8 waves; block handles 128 q-rows of one (b,h); wave owns 16 q-rows.
__global__ __launch_bounds__(256, 1)
void attn_kernel(const _Float16* __restrict__ Q,
                 const _Float16* __restrict__ Kc,
                 const _Float16* __restrict__ VT,
                 _Float16* __restrict__ ctx) {
    const int lane = threadIdx.x & 31;
    const int wave = threadIdx.x >> 5;
    const int l15  = lane & 15;
    const int koff = (lane < 16) ? 0 : 8;
    const int mloc = (lane >> 4) << 3;

    const int nqblk = Sn / 128;
    const int bh    = blockIdx.x / nqblk;
    const int qblk  = blockIdx.x % nqblk;
    const int b = bh / Hn, h = bh % Hn;
    const int q0 = qblk * 128 + wave * 16;

    const _Float16* Qb = Q  + ((size_t)bh * Sn) * Dn;
    const _Float16* Kb = Kc + ((size_t)bh * Sn) * Dn;
    const _Float16* Vb = VT + ((size_t)bh * Dn) * Sn;

    // Q as B operand: lane = query column, elements = d
    v16h qf[3];
    {
        const _Float16* qr = Qb + (size_t)(q0 + l15) * Dn + koff;
#pragma unroll
        for (int i = 0; i < 3; ++i) qf[i] = frag2(qr + i * 32);
    }

    v8f vzero = {};
    v8f acc[6];
#pragma unroll
    for (int t = 0; t < 6; ++t) acc[t] = vzero;
    float mcur = -1e30f, lcur = 0.0f;

    // steady state: no masking needed while kv0+31 < q0
    const int kvFull = q0 & ~31;
    for (int kv0 = 0; kv0 < kvFull; kv0 += 32)
        attn_step<false>(kv0, q0, lane, Kb, Vb, qf, acc, mcur, lcur);
    // single diagonal (masked) step
    attn_step<true>(kvFull, q0, lane, Kb, Vb, qf, acc, mcur, lcur);

    // ---- normalize (broadcast per-query 1/l into C-row domain) and store ctx
    float linv = 1.0f / lcur;
    float lbc[8];
#pragma unroll
    for (int r = 0; r < 8; ++r)
        lbc[r] = __shfl(linv, r + mloc, 32);
#pragma unroll
    for (int t = 0; t < 6; ++t)
#pragma unroll
        for (int r = 0; r < 8; ++r) {
            int s = q0 + r + mloc;
            int d = t*16 + l15;
            ctx[((size_t)(b*Sn + s)) * HIDn + h*Dn + d] = (_Float16)(acc[t][r] * lbc[r]);
        }
}

// ----------------------------------------------------------------- launcher
extern "C" void kernel_launch(void* const* d_in, const int* in_sizes, int n_in,
                              void* d_out, int out_size, void* d_ws, size_t ws_size,
                              hipStream_t stream) {
    const float* hidden = (const float*)d_in[0];
    const float* cosb   = (const float*)d_in[1];
    const float* sinb   = (const float*)d_in[2];
    const float* w_qkv  = (const float*)d_in[3];
    const float* w_o    = (const float*)d_in[4];
    // d_in[5] position_ids / d_in[6] cache_position unused: cos/sin tables are
    // already expanded per (b,s,d) by the reference setup.
    (void)in_sizes; (void)n_in; (void)out_size; (void)ws_size;

    char* ws = (char*)d_ws;
    size_t off = 0;
    auto alloc = [&](size_t bytes) {
        char* p = ws + off;
        off += (bytes + 255) & ~(size_t)255;
        return p;
    };
    _Float16* hx    = (_Float16*)alloc((size_t)Mtok * HIDn * 2);   // 24 MiB
    _Float16* wqkvh = (_Float16*)alloc((size_t)NQKV * HIDn * 2);   // 54 MiB
    _Float16* woh   = (_Float16*)alloc((size_t)HIDn * HIDn * 2);   // 18 MiB
    _Float16* qkvr  = (_Float16*)alloc((size_t)Mtok * NQKV * 2);   // 72 MiB
    _Float16* Qh    = (_Float16*)alloc((size_t)Bn*Hn*Sn*Dn * 2);   // 24 MiB
    _Float16* Kh    = (_Float16*)alloc((size_t)Bn*Hn*Sn*Dn * 2);   // 24 MiB
    _Float16* VTh   = (_Float16*)alloc((size_t)Bn*Hn*Sn*Dn * 2);   // 24 MiB
    _Float16* ctxh  = hx;   // hidden f16 is dead after the QKV GEMM; reuse it

    // 1) precision conversion (f32 -> f16)
    cvt_kernel<<<2048, 256, 0, stream>>>(hidden, hx,    (long long)Mtok * HIDn);
    cvt_kernel<<<2048, 256, 0, stream>>>(w_qkv,  wqkvh, (long long)NQKV * HIDn);
    cvt_kernel<<<2048, 256, 0, stream>>>(w_o,    woh,   (long long)HIDn * HIDn);

    // 2) QKV projection: qkv_raw[4096, 9216]
    gemm_nt_kernel<0><<<dim3(NQKV/128, Mtok/128), 256, 0, stream>>>(
        hx, wqkvh, qkvr, Mtok, NQKV, HIDn);

    // 3) RoPE + reshape into attention layouts (Q pre-scaled by 1/sqrt(D))
    rope_kernel<<<(Bn*Sn*Hn*Dn)/256, 256, 0, stream>>>(qkvr, cosb, sinb, Qh, Kh, VTh);

    // 4) causal flash attention -> ctx[4096, 3072]
    attn_kernel<<<Bn*Hn*(Sn/128), 256, 0, stream>>>(Qh, Kh, VTh, ctxh);

    // 5) output projection -> d_out f32
    gemm_nt_kernel<1><<<dim3(HIDn/128, Mtok/128), 256, 0, stream>>>(
        ctxh, woh, d_out, Mtok, HIDn, HIDn);
}
